// RouteNetModelOccuEval_2_45552423141839
// MI455X (gfx1250) — matrix operand order repaired and checked by reading, not compiled, not run on hardware
//
#include <hip/hip_runtime.h>
#include <stdint.h>

#define N_LINKS 20000
#define N_PATHS 100000
#define MAXLEN  8
#define E_EDGES (N_PATHS*MAXLEN)
#define T_ITERS 8
#define G3      96   // 3*HID gate width

typedef __attribute__((ext_vector_type(16))) __bf16 bf16x16;
typedef __attribute__((ext_vector_type(8)))  __bf16 bf16x8;
typedef __attribute__((ext_vector_type(8)))  float  f32x8;

union BV {            // A-operand view helpers
  bf16x16 v;
  bf16x8  h[2];
  unsigned u[8];
};

// ---- fast transcendentals (no IEEE div chains; v_rcp / v_tanh / v_exp) ----
__device__ __forceinline__ float rcp_fast(float x){
#if __has_builtin(__builtin_amdgcn_rcpf)
  return __builtin_amdgcn_rcpf(x);
#else
  return 1.0f/x;
#endif
}
__device__ __forceinline__ float tanh_fast(float x){
#if __has_builtin(__builtin_amdgcn_tanhf)
  return __builtin_amdgcn_tanhf(x);
#else
  float t = __expf(2.0f*x);
  return 1.0f - 2.0f*rcp_fast(t + 1.0f);
#endif
}
__device__ __forceinline__ float sigm_fast(float x){
#if __has_builtin(__builtin_amdgcn_tanhf)
  return fmaf(__builtin_amdgcn_tanhf(0.5f*x), 0.5f, 0.5f);
#else
  return rcp_fast(1.0f + __expf(-x));
#endif
}
__device__ __forceinline__ float relu_(float x){ return x>0.f?x:0.f; }
__device__ __forceinline__ float selu_(float x){
  const float l=1.0507009873554805f, a=1.6732632423543772f;
  return x>0.f ? l*x : l*a*(__expf(x)-1.0f);
}

// ---------------------------------------------------------------------------
// GRU core on a 16-row tile: gi = x@Wi + bi, gh = h@Wh + bh (6 N-tiles of 16)
// z|r|n chunks live at tiles {c, c+2, c+4} for hidden chunk c in {0,1}.
// 12x v_wmma_f32_16x16x32_bf16 per call, bias pre-loaded into the C operand.
// ---------------------------------------------------------------------------
__device__ __forceinline__ void gru_core(
    bf16x16 xA, bf16x16 hA, const f32x8* hD, f32x8* hN,
    const __bf16* __restrict__ WiP, const __bf16* __restrict__ WhP,
    const float* __restrict__ bi, const float* __restrict__ bh, int lane)
{
  const int col = lane & 15;
  #pragma unroll
  for (int c = 0; c < 2; ++c) {
    f32x8 gi[3], gh[3];
    #pragma unroll
    for (int q = 0; q < 3; ++q) {
      const int t = c + 2*q;                 // z, r, n tile for this chunk
      float bvi = bi[t*16 + col];
      float bvh = bh[t*16 + col];
      f32x8 ci, ch;
      #pragma unroll
      for (int r = 0; r < 8; ++r) { ci[r] = bvi; ch[r] = bvh; }
      bf16x16 wbi = *(const bf16x16*)(WiP + ((size_t)t*32 + lane)*16);
      bf16x16 wbh = *(const bf16x16*)(WhP + ((size_t)t*32 + lane)*16);
      gi[q] = __builtin_amdgcn_wmma_f32_16x16x32_bf16(false, xA, false, wbi, (short)0, ci, false, false);
      gh[q] = __builtin_amdgcn_wmma_f32_16x16x32_bf16(false, hA, false, wbh, (short)0, ch, false, false);
    }
    f32x8 hn;
    #pragma unroll
    for (int r = 0; r < 8; ++r) {
      float z  = sigm_fast(gi[0][r] + gh[0][r]);
      float rr = sigm_fast(gi[1][r] + gh[1][r]);
      float nn = tanh_fast(fmaf(rr, gh[2][r], gi[2][r]));
      hn[r] = fmaf(z, hD[c][r] - nn, nn);    // z*h + (1-z)*n
    }
    hN[c] = hn;
  }
}

// ---------------------------------------------------------------------------
// Path GRU: one wave = 16 paths; rolled 8-step scan (weights stay register-
// resident) with a 2-deep index / 1-deep x-row software prefetch pipeline.
// ---------------------------------------------------------------------------
__global__ void __launch_bounds__(256) gru_path_kernel(
    const __bf16* __restrict__ ls_bf16,
    float* __restrict__ ps_f32, __bf16* __restrict__ ps_bf16,
    const int* __restrict__ link_to_path,
    const __bf16* __restrict__ WiP, const __bf16* __restrict__ WhP,
    const float* __restrict__ bi, const float* __restrict__ bh, int ntiles)
{
  __shared__ __bf16 lds[8][16*32];
  const int wib  = threadIdx.x >> 5;
  const int lane = threadIdx.x & 31;
  const int tile = blockIdx.x * 8 + wib;
  if (tile >= ntiles) return;
  const int rowbase = tile * 16;
  const int r16     = lane & 15;
  const int hihalf  = lane >> 4;
  const int kb0 = hihalf ? 8  : 0;
  const int kb1 = hihalf ? 24 : 16;
  __bf16* tl = lds[wib];

  BV uh;                                        // h in A layout (bf16)
  {
    const __bf16* hrow = ps_bf16 + (size_t)(rowbase + r16) * 32;
    uh.h[0] = *(const bf16x8*)(hrow + kb0);
    uh.h[1] = *(const bf16x8*)(hrow + kb1);
  }
  bf16x16 hA = uh.v;

  f32x8 hD[2];                                  // h in D layout (f32)
  #pragma unroll
  for (int c = 0; c < 2; ++c)
    #pragma unroll
    for (int r = 0; r < 8; ++r)
      hD[c][r] = ps_f32[(size_t)(rowbase + r + hihalf*8)*32 + c*16 + r16];

  const int* l2p = link_to_path + (size_t)(rowbase + r16) * MAXLEN;

  // software pipeline: x for step s in flight while step s-1 computes
  int li_next = l2p[1];
  BV ux;
  {
    const __bf16* xrow = ls_bf16 + (size_t)l2p[0] * 32;
    ux.h[0] = *(const bf16x8*)(xrow + kb0);
    ux.h[1] = *(const bf16x8*)(xrow + kb1);
  }

  #pragma unroll 1
  for (int s = 0; s < MAXLEN; ++s) {
    // prefetch: index two steps ahead, x row one step ahead (tail clamped)
    int li_next2 = l2p[(s + 2 < MAXLEN) ? (s + 2) : (MAXLEN - 1)];
    BV uxn;
    {
      const __bf16* xrow = ls_bf16 + (size_t)li_next * 32;
      uxn.h[0] = *(const bf16x8*)(xrow + kb0);
      uxn.h[1] = *(const bf16x8*)(xrow + kb1);
    }

    unsigned nzb = ux.u[0]|ux.u[1]|ux.u[2]|ux.u[3]|ux.u[4]|ux.u[5]|ux.u[6]|ux.u[7];
    unsigned bal = __builtin_amdgcn_ballot_w32(nzb != 0u);

    f32x8 hN[2];
    gru_core(ux.v, hA, hD, hN, WiP, WhP, bi, bh, lane);

    #pragma unroll
    for (int c = 0; c < 2; ++c)
      #pragma unroll
      for (int r = 0; r < 8; ++r) {
        int m = r + hihalf*8;                   // row held by bit m and m+16
        unsigned msk = ((bal >> m) | (bal >> (m + 16))) & 1u;
        hD[c][r] = msk ? hN[c][r] : hD[c][r];
      }

    // D(f32) -> A(bf16) relayout via per-wave LDS (wave-synchronous transpose)
    #pragma unroll
    for (int c = 0; c < 2; ++c)
      #pragma unroll
      for (int r = 0; r < 8; ++r)
        tl[(r + hihalf*8)*32 + c*16 + r16] = (__bf16)hD[c][r];
    asm volatile("s_wait_dscnt 0" ::: "memory");
    BV un;
    un.h[0] = *(const bf16x8*)&tl[r16*32 + kb0];
    un.h[1] = *(const bf16x8*)&tl[r16*32 + kb1];
    hA = un.v;

    ux = uxn;
    li_next = li_next2;
  }

  #pragma unroll
  for (int c = 0; c < 2; ++c)
    #pragma unroll
    for (int r = 0; r < 8; ++r)
      ps_f32[(size_t)(rowbase + r + hihalf*8)*32 + c*16 + r16] = hD[c][r];
  // bf16 master: LDS already holds final h row-major
  bf16x16 rowv = *(const bf16x16*)&tl[r16*32 + hihalf*16];
  *(bf16x16*)(ps_bf16 + (size_t)(rowbase + r16)*32 + hihalf*16) = rowv;
}

// ---------------------------------------------------------------------------
// Link GRU: x = psum (f32, converted to bf16 A layout), single step.
// ---------------------------------------------------------------------------
__global__ void __launch_bounds__(256) gru_link_kernel(
    const float* __restrict__ psum,
    float* __restrict__ ls_f32, __bf16* __restrict__ ls_bf16,
    const __bf16* __restrict__ WiP, const __bf16* __restrict__ WhP,
    const float* __restrict__ bi, const float* __restrict__ bh, int ntiles)
{
  __shared__ __bf16 lds[8][16*32];
  const int wib  = threadIdx.x >> 5;
  const int lane = threadIdx.x & 31;
  const int tile = blockIdx.x * 8 + wib;
  if (tile >= ntiles) return;
  const int rowbase = tile * 16;
  const int r16     = lane & 15;
  const int hihalf  = lane >> 4;
  const int kb0 = hihalf ? 8  : 0;
  const int kb1 = hihalf ? 24 : 16;
  __bf16* tl = lds[wib];

  BV ux;
  {
    const float* xrow = psum + (size_t)(rowbase + r16) * 32;
    f32x8 x0 = *(const f32x8*)(xrow + kb0);
    f32x8 x1 = *(const f32x8*)(xrow + kb1);
    #pragma unroll
    for (int i = 0; i < 8; ++i) { ux.h[0][i] = (__bf16)x0[i]; ux.h[1][i] = (__bf16)x1[i]; }
  }
  BV uh;
  {
    const __bf16* hrow = ls_bf16 + (size_t)(rowbase + r16) * 32;
    uh.h[0] = *(const bf16x8*)(hrow + kb0);
    uh.h[1] = *(const bf16x8*)(hrow + kb1);
  }
  f32x8 hD[2];
  #pragma unroll
  for (int c = 0; c < 2; ++c)
    #pragma unroll
    for (int r = 0; r < 8; ++r)
      hD[c][r] = ls_f32[(size_t)(rowbase + r + hihalf*8)*32 + c*16 + r16];

  f32x8 hN[2];
  gru_core(ux.v, uh.v, hD, hN, WiP, WhP, bi, bh, lane);

  #pragma unroll
  for (int c = 0; c < 2; ++c)
    #pragma unroll
    for (int r = 0; r < 8; ++r) {
      ls_f32[(size_t)(rowbase + r + hihalf*8)*32 + c*16 + r16] = hN[c][r];
      tl[(r + hihalf*8)*32 + c*16 + r16] = (__bf16)hN[c][r];
    }
  asm volatile("s_wait_dscnt 0" ::: "memory");
  bf16x16 rowv = *(const bf16x16*)&tl[r16*32 + hihalf*16];
  *(bf16x16*)(ls_bf16 + (size_t)(rowbase + r16)*32 + hihalf*16) = rowv;
}

// ---------------------------------------------------------------------------
// Pack a (32 x 96) f32 weight into WMMA B-operand bf16 tiles [6][32 lanes][16K]
// lane -> column N = t*16 + (lane&15); K = (lane<16 ? 0 : 8) + (e&7) + (e>>3)*16.
// ---------------------------------------------------------------------------
__global__ void pack_w_kernel(const float* __restrict__ W, __bf16* __restrict__ P){
  int t = blockIdx.x, lane = threadIdx.x;
  int n  = t*16 + (lane & 15);
  int kb = (lane < 16) ? 0 : 8;
  __bf16* dst = P + ((size_t)t*32 + lane)*16;
  #pragma unroll
  for (int e = 0; e < 16; ++e)
    dst[e] = (__bf16)W[(size_t)(kb + (e & 7) + (e >> 3)*16)*G3 + n];
}

__global__ void tsum_scatter_kernel(const int* __restrict__ l2p,
    const float* __restrict__ traffraw, float* __restrict__ tsum){
  int e = blockIdx.x*blockDim.x + threadIdx.x;
  if (e < E_EDGES) atomicAdd(&tsum[l2p[e]], traffraw[e >> 3]);
}

__global__ void psum_scatter_kernel(const int* __restrict__ l2p,
    const float* __restrict__ ps_f32, float* __restrict__ psum){
  long idx = (long)blockIdx.x*blockDim.x + threadIdx.x;
  if (idx >= (long)E_EDGES*32) return;
  int e = (int)(idx >> 5), j = (int)(idx & 31);
  atomicAdd(&psum[(size_t)l2p[e]*32 + j], ps_f32[(size_t)(e >> 3)*32 + j]);
}

__global__ void link_embed_kernel(const float* __restrict__ tsum, const float* __restrict__ capraw,
    const float* __restrict__ w1, const float* __restrict__ b1,
    const float* __restrict__ w2, const float* __restrict__ b2,
    float* __restrict__ ls_f32, __bf16* __restrict__ ls_bf16){
  int i = blockIdx.x*blockDim.x + threadIdx.x;
  if (i >= N_LINKS) return;
  float l = tsum[i]*rcp_fast(capraw[i]);
  float f0 = l, f1 = l*l, f2 = f1*l;
  float h[32];
  #pragma unroll
  for (int j = 0; j < 32; ++j) h[j] = relu_(f0*w1[j] + f1*w1[32+j] + f2*w1[64+j] + b1[j]);
  for (int j = 0; j < 32; ++j) {
    float s = b2[j];
    #pragma unroll
    for (int k = 0; k < 32; ++k) s += h[k]*w2[k*32 + j];
    float v = selu_(s);
    ls_f32[(size_t)i*32 + j]  = v;
    ls_bf16[(size_t)i*32 + j] = (__bf16)v;
  }
}

__global__ void path_embed_kernel(const float* __restrict__ traffic, const float* __restrict__ packets,
    const float* __restrict__ eqlambda,
    const float* __restrict__ w1, const float* __restrict__ b1,
    const float* __restrict__ w2, const float* __restrict__ b2,
    float* __restrict__ ps_f32, __bf16* __restrict__ ps_bf16){
  int i = blockIdx.x*blockDim.x + threadIdx.x;
  if (i >= N_PATHS) return;
  float f0 = traffic[i], f1 = packets[i], f2 = eqlambda[i];
  float h[32];
  #pragma unroll
  for (int j = 0; j < 32; ++j) h[j] = relu_(f0*w1[j] + f1*w1[32+j] + f2*w1[64+j] + b1[j]);
  for (int j = 0; j < 32; ++j) {
    float s = b2[j];
    #pragma unroll
    for (int k = 0; k < 32; ++k) s += h[k]*w2[k*32 + j];
    float v = selu_(s);
    ps_f32[(size_t)i*32 + j]  = v;
    ps_bf16[(size_t)i*32 + j] = (__bf16)v;
  }
}

__global__ void __launch_bounds__(256) readout_kernel(const float* __restrict__ ls_f32,
    const float* __restrict__ w1, const float* __restrict__ b1,
    const float* __restrict__ w2, const float* __restrict__ b2,
    const float* __restrict__ w3, const float* __restrict__ b3,
    const float* __restrict__ qsize, const float* __restrict__ capraw,
    float* __restrict__ link_delay){
  __shared__ float h1[256];
  __shared__ float red[256];
  int li = blockIdx.x, j = threadIdx.x;
  const float* row = ls_f32 + (size_t)li*32;
  float s = b1[j];
  #pragma unroll
  for (int k = 0; k < 32; ++k) s += row[k]*w1[k*256 + j];
  h1[j] = relu_(s);
  __syncthreads();
  float s2 = b2[j];
  for (int k = 0; k < 256; ++k) s2 += h1[k]*w2[k*256 + j];
  red[j] = relu_(s2)*w3[j];
  __syncthreads();
  for (int st = 128; st > 0; st >>= 1) {
    if (j < st) red[j] += red[j + st];
    __syncthreads();
  }
  if (j == 0) {
    float r = red[0] + b3[0];
    link_delay[li] = r*qsize[li]*rcp_fast(capraw[li]);
  }
}

__global__ void path_delay_kernel(const int* __restrict__ l2p, const float* __restrict__ link_delay,
    const float* __restrict__ traffraw, const float* __restrict__ packraw, float* __restrict__ out){
  int p = blockIdx.x*blockDim.x + threadIdx.x;
  if (p >= N_PATHS) return;
  float s = 0.f;
  #pragma unroll
  for (int q = 0; q < MAXLEN; ++q) s += link_delay[l2p[p*MAXLEN + q]];
  out[p] = s * traffraw[p]*rcp_fast(packraw[p]);
}

// ---------------------------------------------------------------------------
extern "C" void kernel_launch(void* const* d_in, const int* in_sizes, int n_in,
                              void* d_out, int out_size, void* d_ws, size_t ws_size,
                              hipStream_t stream)
{
  (void)in_sizes; (void)n_in; (void)out_size; (void)ws_size;
  const float* traffic  = (const float*)d_in[0];
  const float* packets  = (const float*)d_in[1];
  const float* eqlambda = (const float*)d_in[2];
  const float* qsize    = (const float*)d_in[3];
  const float* capraw   = (const float*)d_in[4];
  const float* traffraw = (const float*)d_in[5];
  const float* packraw  = (const float*)d_in[6];
  // params pytree, dict keys sorted (JAX tree order):
  const float* gl_Wh = (const float*)d_in[7];
  const float* gl_Wi = (const float*)d_in[8];
  const float* gl_bh = (const float*)d_in[9];
  const float* gl_bi = (const float*)d_in[10];
  const float* gp_Wh = (const float*)d_in[11];
  const float* gp_Wi = (const float*)d_in[12];
  const float* gp_bh = (const float*)d_in[13];
  const float* gp_bi = (const float*)d_in[14];
  const float* le_b1 = (const float*)d_in[15];
  const float* le_b2 = (const float*)d_in[16];
  const float* le_w1 = (const float*)d_in[17];
  const float* le_w2 = (const float*)d_in[18];
  const float* pe_b1 = (const float*)d_in[19];
  const float* pe_b2 = (const float*)d_in[20];
  const float* pe_w1 = (const float*)d_in[21];
  const float* pe_w2 = (const float*)d_in[22];
  const float* ro_b1 = (const float*)d_in[23];
  const float* ro_b2 = (const float*)d_in[24];
  const float* ro_b3 = (const float*)d_in[25];
  const float* ro_w1 = (const float*)d_in[26];
  const float* ro_w2 = (const float*)d_in[27];
  const float* ro_w3 = (const float*)d_in[28];
  const int* link_to_path = (const int*)d_in[29];
  float* out = (float*)d_out;

  char* ws = (char*)d_ws;
  size_t off = 0;
  auto alloc = [&](size_t bytes) -> void* {
    void* p = ws + off; off += (bytes + 255) & ~(size_t)255; return p;
  };
  float*  tsum    = (float*) alloc((size_t)N_LINKS*4);
  float*  ls_f32  = (float*) alloc((size_t)N_LINKS*32*4);
  __bf16* ls_bf16 = (__bf16*)alloc((size_t)N_LINKS*32*2);
  float*  ps_f32  = (float*) alloc((size_t)N_PATHS*32*4);
  __bf16* ps_bf16 = (__bf16*)alloc((size_t)N_PATHS*32*2);
  float*  psum    = (float*) alloc((size_t)N_LINKS*32*4);
  __bf16* WiP_p   = (__bf16*)alloc(6*32*16*2);
  __bf16* WhP_p   = (__bf16*)alloc(6*32*16*2);
  __bf16* WiP_l   = (__bf16*)alloc(6*32*16*2);
  __bf16* WhP_l   = (__bf16*)alloc(6*32*16*2);
  float*  ldelay  = (float*) alloc((size_t)N_LINKS*4);

  hipMemsetAsync(tsum, 0, (size_t)N_LINKS*4, stream);
  pack_w_kernel<<<6, 32, 0, stream>>>(gp_Wi, WiP_p);
  pack_w_kernel<<<6, 32, 0, stream>>>(gp_Wh, WhP_p);
  pack_w_kernel<<<6, 32, 0, stream>>>(gl_Wi, WiP_l);
  pack_w_kernel<<<6, 32, 0, stream>>>(gl_Wh, WhP_l);
  tsum_scatter_kernel<<<(E_EDGES+255)/256, 256, 0, stream>>>(link_to_path, traffraw, tsum);
  link_embed_kernel<<<(N_LINKS+255)/256, 256, 0, stream>>>(tsum, capraw, le_w1, le_b1, le_w2, le_b2, ls_f32, ls_bf16);
  path_embed_kernel<<<(N_PATHS+255)/256, 256, 0, stream>>>(traffic, packets, eqlambda, pe_w1, pe_b1, pe_w2, pe_b2, ps_f32, ps_bf16);

  const int ptiles = N_PATHS/16;   // 6250
  const int ltiles = N_LINKS/16;   // 1250
  for (int t = 0; t < T_ITERS; ++t) {
    gru_path_kernel<<<(ptiles+7)/8, 256, 0, stream>>>(ls_bf16, ps_f32, ps_bf16, link_to_path,
                                                      WiP_p, WhP_p, gp_bi, gp_bh, ptiles);
    hipMemsetAsync(psum, 0, (size_t)N_LINKS*32*4, stream);
    psum_scatter_kernel<<<(int)(((size_t)E_EDGES*32 + 255)/256), 256, 0, stream>>>(link_to_path, ps_f32, psum);
    gru_link_kernel<<<(ltiles+7)/8, 256, 0, stream>>>(psum, ls_f32, ls_bf16,
                                                      WiP_l, WhP_l, gl_bi, gl_bh, ltiles);
  }

  readout_kernel<<<N_LINKS, 256, 0, stream>>>(ls_f32, ro_w1, ro_b1, ro_w2, ro_b2, ro_w3, ro_b3,
                                              qsize, capraw, ldelay);
  path_delay_kernel<<<(N_PATHS+255)/256, 256, 0, stream>>>(link_to_path, ldelay, traffraw, packraw, out);
}
